// EGF_block_11596411700014
// MI455X (gfx1250) — compile-verified, matrix operand order checked by reading, not run on hardware
//
#include <hip/hip_runtime.h>
#include <hip/hip_bf16.h>
#include <math.h>

typedef __attribute__((ext_vector_type(16))) __bf16 v16bf;
typedef __attribute__((ext_vector_type(8)))  __bf16 v8bf;
typedef __attribute__((ext_vector_type(8)))  float  v8f;

#define C_      256
#define HW_     4096
#define NHEAD   8
#define DHEAD   32
#define SEQ     64
#define SCALE_K 0.17677669529663687f   /* 32^-0.5 */
#define LN_EPS  1e-6f
#define LDSP    40                     /* LDS pitch in bf16 (80B, 16B-aligned rows) */

// ---- fragment load from LDS row (A or B^T layout): elems {off..off+7, off+16..off+23}
static __device__ inline v16bf frag_lds(const __bf16* row, int off) {
    const v8bf lo = *reinterpret_cast<const v8bf*>(row + off);
    const v8bf hi = *reinterpret_cast<const v8bf*>(row + off + 16);
    return __builtin_shufflevector(lo, hi, 0,1,2,3,4,5,6,7,8,9,10,11,12,13,14,15);
}

// =====================================================================
// sin/cos table: [4096][32]
// =====================================================================
__global__ __launch_bounds__(256) void sincos_kernel(float* __restrict__ sinb,
                                                     float* __restrict__ cosb) {
    int t = blockIdx.x * 256 + threadIdx.x;      // 131072 exact
    int hw = t >> 5, d = t & 31;
    int j = d >> 1;
    float ang = __expf(-((float)j / 15.0f) * logf(10000.0f));
    float ph = (float)hw * ang;
    sinb[t] = sinf(ph);
    cosb[t] = cosf(ph);
}

// =====================================================================
// GEMM: 128x128 tile, K=256 in steps of 32, bf16 WMMA, f32 accum,
// double-buffered LDS (1 barrier / k-step, loads overlap WMMA).
// MODE 0: A = x (NCHW, transpose fused), 768 cols -> q / k*scale / v (+bias)
// MODE 1: A = Arow + Aadd (out+lepe), W=Wo, epilogue += x_nchw (residual) -> z1
// MODE 2: A = Arow (t1), W=Wfc, epilogue += t1 -> z2
// =====================================================================
template<int MODE>
__global__ __launch_bounds__(256) void gemm_kernel(
    const float* __restrict__ x_nchw,
    const float* Arow, const float* Aadd,
    const float* __restrict__ Wq, const float* __restrict__ Wk, const float* __restrict__ Wv,
    const float* __restrict__ bq, const float* __restrict__ bk, const float* __restrict__ bv,
    float* outq, float* outk, float* outv,
    const float* resid_row, const float* __restrict__ resid_nchw)
{
    const int tid  = threadIdx.x;
    const int lane = tid & 31;
    const int wave = tid >> 5;
    const int wm   = wave & 1;            // M half (64 rows)
    const int wn   = wave >> 1;           // N quarter (32 cols)
    const int m0   = blockIdx.x * 128;

    int sel, c0;
    if (MODE == 0) { sel = blockIdx.y >> 1; c0 = (blockIdx.y & 1) * 128; }
    else           { sel = 0;               c0 = blockIdx.y * 128; }
    const float* W    = (sel == 0) ? Wq : (sel == 1) ? Wk : Wv;
    const float* bias = (sel == 0) ? bq : (sel == 1) ? bk : bv;
    float*       Out  = (sel == 0) ? outq : (sel == 1) ? outk : outv;

    __shared__ __align__(16) __bf16 As[2][128 * LDSP];
    __shared__ __align__(16) __bf16 Bs[2][128 * LDSP];

    const v8f vzero = {};
    v8f acc[4][2];
    for (int mt = 0; mt < 4; ++mt) for (int nt = 0; nt < 2; ++nt) acc[mt][nt] = vzero;

    const int b_idx = m0 >> 12;
    const int hw0   = m0 & 4095;
    const int kb    = (lane < 16) ? 0 : 8;
    const int lm    = lane & 15;
    const float* xb = (MODE == 0) ? (x_nchw + (size_t)b_idx * C_ * HW_) : nullptr;

    // stage tile at k0 into registers (global loads, fp32->bf16)
    auto stage = [&](int k0, __bf16* aR, __bf16* bR) {
        if (MODE == 0) {
            const int i = tid & 127, jh = tid >> 7;
            for (int jj = 0; jj < 16; ++jj)
                aR[jj] = (__bf16)xb[(size_t)(k0 + jj * 2 + jh) * HW_ + hw0 + i];
        } else {
            const int kl = tid & 31, ih = tid >> 5;
            for (int ii = 0; ii < 16; ++ii) {
                size_t idx = (size_t)(m0 + ii * 8 + ih) * C_ + k0 + kl;
                float va = Arow[idx];
                if (MODE == 1) va += Aadd[idx];
                aR[ii] = (__bf16)va;
            }
        }
        const int nl = tid & 127, kh = tid >> 7;
        for (int kk = 0; kk < 16; ++kk)
            bR[kk] = (__bf16)W[(size_t)(k0 + kk * 2 + kh) * C_ + c0 + nl];
        if (k0 + 32 < 256)   // gfx1250 global_prefetch_b8 on next weight tile
            __builtin_prefetch(&W[(size_t)(k0 + 32 + kh) * C_ + c0 + nl], 0, 1);
    };
    // commit staged registers to LDS buffer
    auto commit = [&](const __bf16* aR, const __bf16* bR, int buf) {
        if (MODE == 0) {
            const int i = tid & 127, jh = tid >> 7;
            for (int jj = 0; jj < 16; ++jj)
                As[buf][i * LDSP + jj * 2 + jh] = aR[jj];
        } else {
            const int kl = tid & 31, ih = tid >> 5;
            for (int ii = 0; ii < 16; ++ii)
                As[buf][(ii * 8 + ih) * LDSP + kl] = aR[ii];
        }
        const int nl = tid & 127, kh = tid >> 7;
        for (int kk = 0; kk < 16; ++kk)
            Bs[buf][nl * LDSP + kk * 2 + kh] = bR[kk];
    };

    {   // prologue: tile 0 -> buffer 0
        __bf16 aR[16], bR[16];
        stage(0, aR, bR);
        commit(aR, bR, 0);
    }

    for (int ks = 0; ks < 8; ++ks) {
        __syncthreads();                       // buffer (ks&1) visible
        const int cur = ks & 1;
        __bf16 aR[16], bR[16];
        if (ks < 7) stage(32 * (ks + 1), aR, bR);   // overlap with WMMA below

        v16bf afr[4];
        for (int mt = 0; mt < 4; ++mt)
            afr[mt] = frag_lds(&As[cur][(wm * 64 + mt * 16 + lm) * LDSP], kb);
        for (int nt = 0; nt < 2; ++nt) {
            v16bf bfr = frag_lds(&Bs[cur][(wn * 32 + nt * 16 + lm) * LDSP], kb);
            for (int mt = 0; mt < 4; ++mt)
                acc[mt][nt] = __builtin_amdgcn_wmma_f32_16x16x32_bf16(
                    false, afr[mt], false, bfr, (short)0, acc[mt][nt], false, false);
        }
        if (ks < 7) commit(aR, bR, cur ^ 1);
    }

    // ---- epilogue
    const int rsel = lane >> 4;
    for (int mt = 0; mt < 4; ++mt)
      for (int nt = 0; nt < 2; ++nt)
        for (int r = 0; r < 8; ++r) {
            int row = m0 + wm * 64 + mt * 16 + r + rsel * 8;
            int c   = c0 + wn * 32 + nt * 16 + lm;
            float v = acc[mt][nt][r] + bias[c];
            if (MODE == 0) { if (sel == 1) v *= SCALE_K; }
            if (MODE == 1) v += resid_nchw[((size_t)(row >> 12) * C_ + c) * HW_ + (row & 4095)];
            if (MODE == 2) v += resid_row[(size_t)row * C_ + c];
            Out[(size_t)row * C_ + c] = v;
        }
}

// =====================================================================
// RoPE (theta-shift), in place on q and k (grid.y selects tensor)
// =====================================================================
__global__ __launch_bounds__(256) void rope_kernel(float* q, float* k,
                                                   const float* __restrict__ sinb,
                                                   const float* __restrict__ cosb) {
    size_t t = (size_t)blockIdx.x * 256 + threadIdx.x;   // 8388608 exact
    float* a = (blockIdx.y == 0) ? q : k;
    size_t m = t >> 7;
    int p  = (int)(t & 127);
    int cc = p * 2;
    int hw = (int)(m & 4095);
    int din = cc & 31;
    float s0 = sinb[hw * 32 + din], s1 = sinb[hw * 32 + din + 1];
    float c0 = cosb[hw * 32 + din], c1 = cosb[hw * 32 + din + 1];
    size_t idx = m * C_ + cc;
    float xa = a[idx], xb = a[idx + 1];
    a[idx]     = xa * c0 - xb * s0;     // even: x*cos - x_{+1}*sin
    a[idx + 1] = xb * c1 + xa * s1;     // odd : x*cos + x_{-1}*sin
}

// =====================================================================
// LEPE: 5x5 depthwise conv on v (NHWC), zero pad 2
// =====================================================================
__global__ __launch_bounds__(256) void lepe_kernel(const float* __restrict__ v,
                                                   const float* __restrict__ wgt,
                                                   const float* __restrict__ bias,
                                                   float* __restrict__ out) {
    size_t t = (size_t)blockIdx.x * 256 + threadIdx.x;   // 16777216 exact
    int c = (int)(t & 255);
    size_t m = t >> 8;
    int w = (int)(m & 63), h = (int)((m >> 6) & 63), b = (int)(m >> 12);
    float acc = bias[c];
    for (int kh = 0; kh < 5; ++kh) {
        int hh = h + kh - 2;
        if (hh < 0 || hh > 63) continue;
        for (int kw = 0; kw < 5; ++kw) {
            int ww = w + kw - 2;
            if (ww < 0 || ww > 63) continue;
            acc += v[(((size_t)b << 12) + (hh << 6) + ww) * C_ + c] * wgt[(kh * 5 + kw) * C_ + c];
        }
    }
    out[m * C_ + c] = acc;
}

// =====================================================================
// Axial retention: one wave per (b, line, head). 64x64 scores in regs,
// decay mask + softmax in regs, P via LDS for PV matmul. out may alias vin.
// MODE 0: attend along W (line = h); MODE 1: attend along H (line = w)
// =====================================================================
template<int MODE>
__global__ __launch_bounds__(128) void attn_kernel(const float* __restrict__ qr,
                                                   const float* __restrict__ kr,
                                                   const float* vin, float* outp) {
    const int lane = threadIdx.x & 31;
    const int wave = threadIdx.x >> 5;
    const int inst = blockIdx.x * 4 + wave;
    const int n    = inst & 7;
    const int line = (inst >> 3) & 63;
    const int b    = inst >> 9;
    size_t base, stride;
    if (MODE == 0) { base = (((size_t)b * 64 + line) * 64) * C_ + n * DHEAD; stride = C_; }
    else           { base = ((size_t)b * 64 * 64 + line) * C_ + n * DHEAD;   stride = 64 * C_; }

    const float decay = logf(1.0f - exp2f(-1.0f - 3.0f * (float)n / 8.0f));
    const int kb   = (lane < 16) ? 0 : 8;
    const int lm   = lane & 15;
    const int rsel = lane >> 4;
    const v8f vzero = {};

    // Q fragments (A layout), bf16
    v16bf aq[4];
    for (int mt = 0; mt < 4; ++mt) {
        const float* p = qr + base + (size_t)(mt * 16 + lm) * stride;
        v16bf f;
        for (int u = 0; u < 8; ++u) { f[u] = (__bf16)p[kb + u]; f[u + 8] = (__bf16)p[kb + 16 + u]; }
        aq[mt] = f;
    }
    // S = Qr Kr^T  (K already has D^-1/2 folded in)
    v8f s[4][4];
    for (int nt = 0; nt < 4; ++nt) {
        const float* p = kr + base + (size_t)(nt * 16 + lm) * stride;
        v16bf bf_;
        for (int u = 0; u < 8; ++u) { bf_[u] = (__bf16)p[kb + u]; bf_[u + 8] = (__bf16)p[kb + 16 + u]; }
        for (int mt = 0; mt < 4; ++mt)
            s[mt][nt] = __builtin_amdgcn_wmma_f32_16x16x32_bf16(
                false, aq[mt], false, bf_, (short)0, vzero, false, false);
    }
    // decay mask + row softmax (rows split across 16-lane halves; xor<=8 stays inside)
    for (int mt = 0; mt < 4; ++mt) {
        for (int nt = 0; nt < 4; ++nt)
            for (int r = 0; r < 8; ++r) {
                float qi = (float)(mt * 16 + r + rsel * 8);
                float ki = (float)(nt * 16 + lm);
                s[mt][nt][r] += fabsf(qi - ki) * decay;
            }
        v8f mx = s[mt][0];
        for (int nt = 1; nt < 4; ++nt)
            for (int r = 0; r < 8; ++r) mx[r] = fmaxf(mx[r], s[mt][nt][r]);
        for (int off = 8; off; off >>= 1)
            for (int r = 0; r < 8; ++r) mx[r] = fmaxf(mx[r], __shfl_xor(mx[r], off, 32));
        v8f sum = vzero;
        for (int nt = 0; nt < 4; ++nt)
            for (int r = 0; r < 8; ++r) { s[mt][nt][r] = __expf(s[mt][nt][r] - mx[r]); sum[r] += s[mt][nt][r]; }
        for (int off = 8; off; off >>= 1)
            for (int r = 0; r < 8; ++r) sum[r] += __shfl_xor(sum[r], off, 32);
        for (int r = 0; r < 8; ++r) sum[r] = 1.0f / sum[r];
        for (int nt = 0; nt < 4; ++nt)
            for (int r = 0; r < 8; ++r) s[mt][nt][r] *= sum[r];
    }
    // P -> LDS (re-fragment into A layout)
    __shared__ __align__(16) __bf16 Pt[4][64][72];
    for (int mt = 0; mt < 4; ++mt)
        for (int nt = 0; nt < 4; ++nt)
            for (int r = 0; r < 8; ++r)
                Pt[wave][mt * 16 + r + rsel * 8][nt * 16 + lm] = (__bf16)s[mt][nt][r];
    __syncthreads();

    // V fragments (B layout), columns d
    v16bf vf[2][2];
    for (int nt = 0; nt < 2; ++nt)
        for (int kc = 0; kc < 2; ++kc) {
            const float* p = vin + base + nt * 16 + lm;
            int k0 = kc * 32 + kb;
            v16bf f;
            for (int u = 0; u < 8; ++u) {
                f[u]     = (__bf16)p[(size_t)(k0 + u) * stride];
                f[u + 8] = (__bf16)p[(size_t)(k0 + 16 + u) * stride];
            }
            vf[nt][kc] = f;
        }
    // O = P V
    v8f o[4][2];
    for (int mt = 0; mt < 4; ++mt) {
        const __bf16* prow = &Pt[wave][mt * 16 + lm][0];
        v16bf p0 = frag_lds(prow, kb);
        v16bf p1 = frag_lds(prow + 32, kb);
        for (int nt = 0; nt < 2; ++nt) {
            v8f a = __builtin_amdgcn_wmma_f32_16x16x32_bf16(
                false, p0, false, vf[nt][0], (short)0, vzero, false, false);
            o[mt][nt] = __builtin_amdgcn_wmma_f32_16x16x32_bf16(
                false, p1, false, vf[nt][1], (short)0, a, false, false);
        }
    }
    // store (aliases vin safely: per-block slices disjoint, reads precede stores)
    for (int mt = 0; mt < 4; ++mt)
        for (int nt = 0; nt < 2; ++nt)
            for (int r = 0; r < 8; ++r)
                outp[base + (size_t)(mt * 16 + r + rsel * 8) * stride + nt * 16 + lm] = o[mt][nt][r];
}

// =====================================================================
// LayerNorm over C=256; one wave per row (8 rows per block)
// =====================================================================
__global__ __launch_bounds__(256) void ln_kernel(const float* __restrict__ in,
                                                 const float* __restrict__ g,
                                                 const float* __restrict__ bb,
                                                 float* __restrict__ out) {
    const int lane = threadIdx.x & 31, wave = threadIdx.x >> 5;
    const size_t row = (size_t)blockIdx.x * 8 + wave;
    float v[8], s = 0.f, sq = 0.f;
    for (int u = 0; u < 8; ++u) { v[u] = in[row * C_ + u * 32 + lane]; s += v[u]; sq += v[u] * v[u]; }
    for (int off = 16; off; off >>= 1) { s += __shfl_xor(s, off, 32); sq += __shfl_xor(sq, off, 32); }
    float mean = s * (1.0f / 256.0f);
    float var  = sq * (1.0f / 256.0f) - mean * mean;
    float rs   = rsqrtf(var + LN_EPS);
    for (int u = 0; u < 8; ++u) {
        int c = u * 32 + lane;
        out[row * C_ + c] = g[c] * (v[u] - mean) * rs + bb[c];
    }
}

// LayerNorm + NHWC -> NCHW transpose into d_out
__global__ __launch_bounds__(256) void ln2t_kernel(const float* __restrict__ in,
                                                   const float* __restrict__ g,
                                                   const float* __restrict__ bb,
                                                   float* __restrict__ out) {
    const int lane = threadIdx.x & 31, wave = threadIdx.x >> 5;
    const size_t row = (size_t)blockIdx.x * 8 + wave;
    const int b = (int)(row >> 12), hw = (int)(row & 4095);
    float v[8], s = 0.f, sq = 0.f;
    for (int u = 0; u < 8; ++u) { v[u] = in[row * C_ + u * 32 + lane]; s += v[u]; sq += v[u] * v[u]; }
    for (int off = 16; off; off >>= 1) { s += __shfl_xor(s, off, 32); sq += __shfl_xor(sq, off, 32); }
    float mean = s * (1.0f / 256.0f);
    float var  = sq * (1.0f / 256.0f) - mean * mean;
    float rs   = rsqrtf(var + LN_EPS);
    for (int u = 0; u < 8; ++u) {
        int c = u * 32 + lane;
        out[((size_t)b * C_ + c) * HW_ + hw] = g[c] * (v[u] - mean) * rs + bb[c];
    }
}

// =====================================================================
extern "C" void kernel_launch(void* const* d_in, const int* in_sizes, int n_in,
                              void* d_out, int out_size, void* d_ws, size_t ws_size,
                              hipStream_t stream) {
    const float* x      = (const float*)d_in[0];
    const float* Wq     = (const float*)d_in[1];
    const float* bq     = (const float*)d_in[2];
    const float* Wk     = (const float*)d_in[3];
    const float* bk     = (const float*)d_in[4];
    const float* Wv     = (const float*)d_in[5];
    const float* bv     = (const float*)d_in[6];
    const float* lepe_w = (const float*)d_in[7];
    const float* lepe_b = (const float*)d_in[8];
    const float* Wo     = (const float*)d_in[9];
    const float* bo     = (const float*)d_in[10];
    const float* Wfc    = (const float*)d_in[11];
    const float* bfc    = (const float*)d_in[12];
    const float* g1     = (const float*)d_in[13];
    const float* b1     = (const float*)d_in[14];
    const float* g2     = (const float*)d_in[15];
    const float* b2     = (const float*)d_in[16];
    float* dout = (float*)d_out;

    const size_t SZ = (size_t)16 * 64 * 64 * 256;   // 16,777,216 floats
    float* ws   = (float*)d_ws;
    float* q    = ws;
    float* k    = ws + SZ;
    float* v    = ws + 2 * SZ;      // also holds v5, then attention out
    float* lepe = ws + 3 * SZ;      // reused as t1 after Wo GEMM
    float* z1   = ws + 4 * SZ;
    float* sinb = ws + 5 * SZ;
    float* cosb = sinb + 131072;
    float* t1   = lepe;             // free after Wo GEMM consumed lepe
    float* z2   = q;                // free after both attention passes

    // 1. sin/cos tables
    sincos_kernel<<<512, 256, 0, stream>>>(sinb, cosb);
    // 2. fused QKV projection (transpose fused, K pre-scaled)
    gemm_kernel<0><<<dim3(512, 6), 256, 0, stream>>>(x, nullptr, nullptr,
        Wq, Wk, Wv, bq, bk, bv, q, k, v, nullptr, nullptr);
    // 3. RoPE in place on q and k
    rope_kernel<<<dim3(32768, 2), 256, 0, stream>>>(q, k, sinb, cosb);
    // 4. depthwise 5x5 lepe on v (before v is overwritten by v5)
    lepe_kernel<<<65536, 256, 0, stream>>>(v, lepe_w, lepe_b, lepe);
    // 5. retention along W, then along H (v buffer reused in place)
    attn_kernel<0><<<2048, 128, 0, stream>>>(q, k, v, v);
    attn_kernel<1><<<2048, 128, 0, stream>>>(q, k, v, v);
    // 6. (out + lepe) @ Wo + bo + x  -> z1
    gemm_kernel<1><<<dim3(512, 2), 256, 0, stream>>>(nullptr, v, lepe,
        Wo, Wo, Wo, bo, bo, bo, z1, z1, z1, nullptr, x);
    // 7. LN1 -> t1
    ln_kernel<<<8192, 256, 0, stream>>>(z1, g1, b1, t1);
    // 8. t1 @ Wfc + bfc + t1 -> z2
    gemm_kernel<2><<<dim3(512, 2), 256, 0, stream>>>(nullptr, t1, nullptr,
        Wfc, Wfc, Wfc, bfc, bfc, bfc, z2, z2, z2, t1, nullptr);
    // 9. LN2 + transpose to NCHW output
    ln2t_kernel<<<8192, 256, 0, stream>>>(z2, g2, b2, dout);

    (void)in_sizes; (void)n_in; (void)out_size; (void)ws_size;
}
